// ZCIA_Transformer_13322988552642
// MI455X (gfx1250) — matrix operand order repaired and checked by reading, not compiled, not run on hardware
//
#include <hip/hip_runtime.h>
#include <hip/hip_bf16.h>

// ---------------------------------------------------------------------------
// MI455X (gfx1250) set-transformer forward pass.
// Compute-bound (~42 GFLOP vs ~2.5MB HBM traffic) -> everything routed through
// v_wmma_f32_16x16x32_f16 (f16 inputs, f32 accumulate).
// ---------------------------------------------------------------------------

typedef __attribute__((ext_vector_type(16))) _Float16 v16h;
typedef __attribute__((ext_vector_type(8)))  _Float16 v8h;
typedef __attribute__((ext_vector_type(4)))  _Float16 v4h;
typedef __attribute__((ext_vector_type(8)))  float    v8f;

// ---- WMMA fragment helpers (ISA 7.12.2 layouts, wave32) -------------------
// A 16x32 f16: lane L holds row M = L%16; element p -> k = 8*(L/16)+p (p<8),
//              k = 16 + 8*(L/16) + (p-8) (p>=8).
__device__ __forceinline__ v16h frag_ld_a(const _Float16* tile, int stride) {
  const int lane = threadIdx.x & 31;
  const int r = lane & 15, h = lane >> 4;
  const _Float16* p = tile + r * stride + h * 8;
  v8h lo = *(const v8h*)(p);        // k = 8h .. 8h+7
  v8h hi = *(const v8h*)(p + 16);   // k = 16+8h .. 16+8h+7
  v16h f;
#pragma unroll
  for (int i = 0; i < 8; ++i) { f[i] = lo[i]; f[i + 8] = hi[i]; }
  return f;
}

// B 32x16 f16 (stored transposed in LDS as [n][k]): lane L holds col N=L%16;
// element p -> k = 8*(L/16)+p (p<8), k = 16 + 8*(L/16) + (p-8) (p>=8).
__device__ __forceinline__ v16h frag_ld_b(const _Float16* tile, int stride) {
  const int lane = threadIdx.x & 31;
  const int n = lane & 15, h = lane >> 4;
  const _Float16* p = tile + n * stride + h * 8;
  v8h lo = *(const v8h*)(p);        // k = 8h .. 8h+7
  v8h hi = *(const v8h*)(p + 16);   // k = 16+8h .. 16+8h+7
  v16h f;
#pragma unroll
  for (int i = 0; i < 8; ++i) { f[i] = lo[i]; f[i + 8] = hi[i]; }
  return f;
}

__device__ __forceinline__ v8f wmma_f16(v16h a, v16h b, v8f c) {
  return __builtin_amdgcn_wmma_f32_16x16x32_f16(false, a, false, b,
                                                (short)0, c, false, false);
}

__device__ __forceinline__ v4h cvt4(float4 v) {
  v4h h;
  h[0] = (_Float16)v.x; h[1] = (_Float16)v.y;
  h[2] = (_Float16)v.z; h[3] = (_Float16)v.w;
  return h;
}

// ---------------------------------------------------------------------------
// Generic tiled WMMA GEMM.  D = act(scale*(A@B) + bias + mask + Cadd)
//   A: (M,K) fp32, row stride lda, per-z offset zb*aSb + zh*aSh
//   B: fp32; btrans=0 -> (K,N) row-major ldb; btrans=1 -> (N,K) row-major ldb
//   bias_mode: 0 none, 1 per-column, 2 scalar (bias[0])
//   nmask: optional key-padding mask over columns (adds -1e9)
//   zsplit: blockIdx.z -> (zb = z>>3, zh = z&7) else (zb = z, zh = 0)
// Assumes: M % 128 == 0 handled by grid, K % 32 == 0, N % 4 == 0,
// all fp32 sources 16B-aligned with lda/ldb % 4 == 0.
// ---------------------------------------------------------------------------
#define GBM 128
#define GBN 64
#define GBK 32
#define GAST 40   // LDS row stride in halfs (mult of 8 -> 16B-aligned b128 loads)

__global__ void __launch_bounds__(256)
gemm_kernel(const float* __restrict__ Abase, int lda, long aSb, long aSh,
            const float* __restrict__ Bbase, int ldb, long bSb, long bSh, int btrans,
            const float* __restrict__ bias, int bias_mode,
            const float* __restrict__ Cadd, long cSb,
            float* __restrict__ Dbase, int ldd, long dSb, long dSh,
            const unsigned char* __restrict__ nmask, long mSb,
            int M, int N, int K, float scale, int relu, int zsplit)
{
  __shared__ __align__(16) _Float16 la[GBM * GAST];
  __shared__ __align__(16) _Float16 lb[GBN * GAST];

  const int t = threadIdx.x;
  const int wave = t >> 5;
  const int lane = t & 31;
  const int ln16 = lane & 15, lh = lane >> 4;

  int zb = blockIdx.z, zh = 0;
  if (zsplit) { zh = zb & 7; zb >>= 3; }
  const float* A = Abase + zb * aSb + zh * aSh;
  const float* B = Bbase + zb * bSb + zh * bSh;
  float*       D = Dbase + zb * dSb + zh * dSh;
  const unsigned char* msk = nmask ? (nmask + zb * mSb) : nullptr;

  const int m0 = blockIdx.y * GBM;
  const int n0 = blockIdx.x * GBN;

  v8f acc[4];
#pragma unroll
  for (int i = 0; i < 4; ++i)
#pragma unroll
    for (int e = 0; e < 8; ++e) acc[i][e] = 0.f;

  for (int k0 = 0; k0 < K; k0 += GBK) {
    // ---- stage A tile (128x32): 1024 float4 slots, 4 per thread -----------
    {
      float4 av[4];
      int ar[4], ac[4];
#pragma unroll
      for (int q = 0; q < 4; ++q) {              // batch loads (stay in flight)
        int idx = t + q * 256;
        ar[q] = idx >> 3;                        // 8 float4 per 32-wide row
        ac[q] = (idx & 7) * 4;
        av[q] = *(const float4*)(A + (m0 + ar[q]) * lda + k0 + ac[q]);
      }
#pragma unroll
      for (int q = 0; q < 4; ++q)
        *(v4h*)(&la[ar[q] * GAST + ac[q]]) = cvt4(av[q]);
    }
    // ---- stage B tile transposed into [n][k] ------------------------------
    if (btrans) {                                // source rows are (n, k)
      float4 bv[2];
      int br[2], bc[2];
#pragma unroll
      for (int q = 0; q < 2; ++q) {
        int idx = t + q * 256;
        br[q] = idx >> 3;
        bc[q] = (idx & 7) * 4;
        float4 v = {0.f, 0.f, 0.f, 0.f};
        if (n0 + br[q] < N) v = *(const float4*)(B + (n0 + br[q]) * ldb + k0 + bc[q]);
        bv[q] = v;
      }
#pragma unroll
      for (int q = 0; q < 2; ++q)
        *(v4h*)(&lb[br[q] * GAST + bc[q]]) = cvt4(bv[q]);
    } else {                                     // source rows are (k, n)
      float4 bv[2];
      int bk[2], bn[2];
#pragma unroll
      for (int q = 0; q < 2; ++q) {
        int idx = t + q * 256;
        bk[q] = idx >> 4;                        // 16 float4 per 64-wide row
        bn[q] = (idx & 15) * 4;
        float4 v = {0.f, 0.f, 0.f, 0.f};
        if (n0 + bn[q] < N) v = *(const float4*)(B + (k0 + bk[q]) * ldb + n0 + bn[q]);
        bv[q] = v;
      }
#pragma unroll
      for (int q = 0; q < 2; ++q) {              // scatter into [n][k]
        lb[(bn[q] + 0) * GAST + bk[q]] = (_Float16)bv[q].x;
        lb[(bn[q] + 1) * GAST + bk[q]] = (_Float16)bv[q].y;
        lb[(bn[q] + 2) * GAST + bk[q]] = (_Float16)bv[q].z;
        lb[(bn[q] + 3) * GAST + bk[q]] = (_Float16)bv[q].w;
      }
    }
    __syncthreads();
    v16h af = frag_ld_a(la + wave * 16 * GAST, GAST);
    v16h bf[4];
#pragma unroll
    for (int nt = 0; nt < 4; ++nt)               // prefetch all B fragments
      bf[nt] = frag_ld_b(lb + nt * 16 * GAST, GAST);
#pragma unroll
    for (int nt = 0; nt < 4; ++nt)
      acc[nt] = wmma_f16(af, bf[nt], acc[nt]);
    __syncthreads();
  }

  const float bsc = (bias_mode == 2 && bias) ? bias[0] : 0.f;
#pragma unroll
  for (int nt = 0; nt < 4; ++nt) {
    int gn = n0 + nt * 16 + ln16;
    if (gn >= N) continue;
    float badd = bsc;
    if (bias_mode == 1) badd = bias[gn];
    if (msk && msk[gn]) badd += -1e9f;
#pragma unroll
    for (int r = 0; r < 8; ++r) {
      int gm = m0 + wave * 16 + r + lh * 8;      // D: m = reg + 8*half
      float v = acc[nt][r] * scale + badd;
      if (Cadd) v += Cadd[zb * cSb + gm * N + gn];
      if (relu) v = v > 0.f ? v : 0.f;
      D[gm * ldd + gn] = v;
    }
  }
}

// ---------------------------------------------------------------------------
// Fused set-encoder: one block per set i (0..1023).
//   h1[j][e1] = relu(x*w1[0]+m*w1[1]+b1) computed on the fly per K-slab,
//   h2 = relu(h1 @ enc_w2 + b2) via WMMA, column-mean pooled,
//   col = LN(pooled @ post_w + post_b).
// Deterministic column sums (no float atomics).
// ---------------------------------------------------------------------------
__global__ void __launch_bounds__(256)
embed_kernel(const float* __restrict__ x, const float* __restrict__ mm,
             const float* __restrict__ w1, const float* __restrict__ b1,
             const float* __restrict__ w2, const float* __restrict__ b2,
             const float* __restrict__ pw, const float* __restrict__ pb,
             const float* __restrict__ g0, const float* __restrict__ be0,
             float* __restrict__ z)
{
  __shared__ float xv[256], mv[256];
  __shared__ float w1a[256], w1b[256], b1v[256];
  __shared__ __align__(16) _Float16 la[256 * GAST];   // 20 KB
  __shared__ __align__(16) _Float16 lb[64 * GAST];
  __shared__ float part[32][64];                      // deterministic partials
  __shared__ float pooled[256];
  __shared__ float red[256], red2[256];

  const int i = blockIdx.x;
  const int t = threadIdx.x;
  const int wave = t >> 5, lane = t & 31, ln16 = lane & 15, lh = lane >> 4;

  // raw reshape: xf[i, :] = x.flat[i*256 : (i+1)*256]  (contiguous!)
  xv[t] = x[i * 256 + t];
  mv[t] = mm[i * 256 + t];
  w1a[t] = w1[t];
  w1b[t] = w1[256 + t];
  b1v[t] = b1[t];
  pooled[t] = 0.f;
  __syncthreads();

  for (int n0 = 0; n0 < 256; n0 += 64) {
    v8f acc[2][4];
#pragma unroll
    for (int a = 0; a < 2; ++a)
#pragma unroll
      for (int b = 0; b < 4; ++b)
#pragma unroll
        for (int e = 0; e < 8; ++e) acc[a][b][e] = 0.f;

    for (int k0 = 0; k0 < 256; k0 += 32) {
      // stage A: 256x32 slab of h1, computed on the fly (LDS/VALU only)
#pragma unroll
      for (int q = 0; q < 8; ++q) {
        int idx = t + q * 256;                    // 2048 v4h slots
        int j = idx >> 3, c4 = (idx & 7) * 4;
        float xj = xv[j], mj = mv[j];
        v4h h;
#pragma unroll
        for (int u = 0; u < 4; ++u) {
          int e1 = k0 + c4 + u;
          float h1 = xj * w1a[e1] + mj * w1b[e1] + b1v[e1];
          h[u] = (_Float16)(h1 > 0.f ? h1 : 0.f);
        }
        *(v4h*)(&la[j * GAST + c4]) = h;
      }
      // stage B: enc_w2 32x64 tile (k,n) -> transposed [n][k]
      {
        float4 bv[2];
        int bk[2], bn[2];
#pragma unroll
        for (int q = 0; q < 2; ++q) {
          int idx = t + q * 256;
          bk[q] = idx >> 4;
          bn[q] = (idx & 15) * 4;
          bv[q] = *(const float4*)(w2 + (k0 + bk[q]) * 256 + n0 + bn[q]);
        }
#pragma unroll
        for (int q = 0; q < 2; ++q) {
          lb[(bn[q] + 0) * GAST + bk[q]] = (_Float16)bv[q].x;
          lb[(bn[q] + 1) * GAST + bk[q]] = (_Float16)bv[q].y;
          lb[(bn[q] + 2) * GAST + bk[q]] = (_Float16)bv[q].z;
          lb[(bn[q] + 3) * GAST + bk[q]] = (_Float16)bv[q].w;
        }
      }
      __syncthreads();
      v16h bf[4];
#pragma unroll
      for (int nt = 0; nt < 4; ++nt)
        bf[nt] = frag_ld_b(lb + nt * 16 * GAST, GAST);
#pragma unroll
      for (int mi = 0; mi < 2; ++mi) {
        v16h af = frag_ld_a(la + (wave + mi * 8) * 16 * GAST, GAST);
#pragma unroll
        for (int nt = 0; nt < 4; ++nt)
          acc[mi][nt] = wmma_f16(af, bf[nt], acc[mi][nt]);
      }
      __syncthreads();
    }
    // relu + column partial sums (8 rows per lane per tile)
#pragma unroll
    for (int nt = 0; nt < 4; ++nt) {
#pragma unroll
      for (int mi = 0; mi < 2; ++mi) {
        int col = nt * 16 + ln16;
        float bb = b2[n0 + col];
        float s = 0.f;
#pragma unroll
        for (int r = 0; r < 8; ++r) {
          float v = acc[mi][nt][r] + bb;
          s += (v > 0.f ? v : 0.f);
        }
        part[(wave * 2 + mi) * 2 + lh][col] = s;
      }
      __syncthreads();
      if (t < 64) {
        float s = 0.f;
        for (int id = 0; id < 32; ++id) s += part[id][t];
        pooled[n0 + t] += s;
      }
      __syncthreads();
    }
  }
  pooled[t] *= (1.0f / 256.0f);            // mean over R=256 set elements
  __syncthreads();

  // post projection: col[e] = sum_k pooled[k] * post_w[k][e] + post_b[e]
  float a = pb[t];
  for (int k = 0; k < 256; ++k) a += pooled[k] * pw[k * 256 + t];

  // LayerNorm over E=256
  red[t] = a; red2[t] = a * a;
  __syncthreads();
  for (int s = 128; s > 0; s >>= 1) {
    if (t < s) { red[t] += red[t + s]; red2[t] += red2[t + s]; }
    __syncthreads();
  }
  float mu  = red[0] * (1.0f / 256.0f);
  float var = red2[0] * (1.0f / 256.0f) - mu * mu;
  float inv = __frsqrt_rn(var + 1e-5f);
  z[i * 256 + t] = (a - mu) * inv * g0[t] + be0[t];
}

// ---------------------------------------------------------------------------
__global__ void __launch_bounds__(256)
ln_kernel(const float* __restrict__ in, const float* __restrict__ g,
          const float* __restrict__ b, float* __restrict__ out)
{
  __shared__ float r1[256], r2[256];
  const int row = blockIdx.x, t = threadIdx.x;
  float v = in[row * 256 + t];
  r1[t] = v; r2[t] = v * v;
  __syncthreads();
  for (int s = 128; s > 0; s >>= 1) {
    if (t < s) { r1[t] += r1[t + s]; r2[t] += r2[t + s]; }
    __syncthreads();
  }
  float mu  = r1[0] * (1.0f / 256.0f);
  float var = r2[0] * (1.0f / 256.0f) - mu * mu;
  float inv = __frsqrt_rn(var + 1e-5f);
  out[row * 256 + t] = (v - mu) * inv * g[t] + b[t];
}

__global__ void __launch_bounds__(128)
softmax_kernel(float* __restrict__ s)
{
  __shared__ float r[128];
  float* p = s + blockIdx.x * 128;
  const int t = threadIdx.x;
  float v = p[t];
  r[t] = v; __syncthreads();
  for (int q = 64; q > 0; q >>= 1) { if (t < q) r[t] = fmaxf(r[t], r[t + q]); __syncthreads(); }
  float mx = r[0]; __syncthreads();
  float e = __expf(v - mx);
  r[t] = e; __syncthreads();
  for (int q = 64; q > 0; q >>= 1) { if (t < q) r[t] += r[t + q]; __syncthreads(); }
  p[t] = e / r[0];
}

// ---------------------------------------------------------------------------
extern "C" void kernel_launch(void* const* d_in, const int* in_sizes, int n_in,
                              void* d_out, int out_size, void* d_ws, size_t ws_size,
                              hipStream_t stream)
{
  (void)in_sizes; (void)n_in; (void)out_size; (void)ws_size;
  const float* x      = (const float*)d_in[0];
  const float* mIn    = (const float*)d_in[1];
  const unsigned char* pad = (const unsigned char*)d_in[2];
  const float* enc_w1 = (const float*)d_in[3];
  const float* enc_b1 = (const float*)d_in[4];
  const float* enc_w2 = (const float*)d_in[5];
  const float* enc_b2 = (const float*)d_in[6];
  const float* post_w = (const float*)d_in[7];
  const float* post_b = (const float*)d_in[8];
  const float* ln0_g  = (const float*)d_in[9];
  const float* ln0_b  = (const float*)d_in[10];
  const float* wq = (const float*)d_in[11];
  const float* bq = (const float*)d_in[12];
  const float* wk = (const float*)d_in[13];
  const float* bk = (const float*)d_in[14];
  const float* wv = (const float*)d_in[15];
  const float* bv = (const float*)d_in[16];
  const float* wo = (const float*)d_in[17];
  const float* bo = (const float*)d_in[18];
  const float* ln1_g = (const float*)d_in[19];
  const float* ln1_b = (const float*)d_in[20];
  const float* wff1  = (const float*)d_in[21];
  const float* bff1  = (const float*)d_in[22];
  const float* wff2  = (const float*)d_in[23];
  const float* bff2  = (const float*)d_in[24];
  const float* ln2_g = (const float*)d_in[25];
  const float* ln2_b = (const float*)d_in[26];
  const float* head_w = (const float*)d_in[27];
  const float* head_b = (const float*)d_in[28];
  float* out = (float*)d_out;

  // workspace layout (floats), total ~15 MB
  float* ws  = (float*)d_ws;
  float* zb  = ws;                  // (1024,256) token states
  float* qb  = zb + 262144;
  float* kb  = qb + 262144;
  float* vb  = kb + 262144;
  float* cb  = vb + 262144;         // attention context
  float* tb  = cb + 262144;         // pre-LN temp
  float* sb  = tb + 262144;         // scores/probs (8,8,128,128)
  float* fb  = sb + 1048576;        // FF hidden (1024,1024)
  float* zwb = fb + 1048576;        // z @ head_w

  embed_kernel<<<1024, 256, 0, stream>>>(x, mIn, enc_w1, enc_b1, enc_w2, enc_b2,
                                         post_w, post_b, ln0_g, ln0_b, zb);

  const float iscale = 0.17677669529663687f;  // 1/sqrt(HD=32)
  for (int l = 0; l < 4; ++l) {
    const float* wql = wq + (long)l * 65536;
    const float* wkl = wk + (long)l * 65536;
    const float* wvl = wv + (long)l * 65536;
    const float* wol = wo + (long)l * 65536;

    gemm_kernel<<<dim3(4, 8, 1), 256, 0, stream>>>(
        zb, 256, 0, 0, wql, 256, 0, 0, 0, bq + l * 256, 1, nullptr, 0,
        qb, 256, 0, 0, nullptr, 0, 1024, 256, 256, 1.f, 0, 0);
    gemm_kernel<<<dim3(4, 8, 1), 256, 0, stream>>>(
        zb, 256, 0, 0, wkl, 256, 0, 0, 0, bk + l * 256, 1, nullptr, 0,
        kb, 256, 0, 0, nullptr, 0, 1024, 256, 256, 1.f, 0, 0);
    gemm_kernel<<<dim3(4, 8, 1), 256, 0, stream>>>(
        zb, 256, 0, 0, wvl, 256, 0, 0, 0, bv + l * 256, 1, nullptr, 0,
        vb, 256, 0, 0, nullptr, 0, 1024, 256, 256, 1.f, 0, 0);

    // scores[b,h] = (q_bh @ k_bh^T)/sqrt(HD) + pad bias   (z = b*8+h)
    gemm_kernel<<<dim3(2, 1, 64), 256, 0, stream>>>(
        qb, 256, 32768, 32, kb, 256, 32768, 32, 1, nullptr, 0, nullptr, 0,
        sb, 128, 131072, 16384, pad, 128, 128, 128, 32, iscale, 0, 1);

    softmax_kernel<<<8192, 128, 0, stream>>>(sb);

    // ctx[b,h] = probs @ v_bh  (M=128, N=32, K=128)
    gemm_kernel<<<dim3(1, 1, 64), 256, 0, stream>>>(
        sb, 128, 131072, 16384, vb, 256, 32768, 32, 0, nullptr, 0, nullptr, 0,
        cb, 256, 32768, 32, nullptr, 0, 128, 32, 128, 1.f, 0, 1);

    // out proj + residual, then LN1
    gemm_kernel<<<dim3(4, 8, 1), 256, 0, stream>>>(
        cb, 256, 0, 0, wol, 256, 0, 0, 0, bo + l * 256, 1, zb, 0,
        tb, 256, 0, 0, nullptr, 0, 1024, 256, 256, 1.f, 0, 0);
    ln_kernel<<<1024, 256, 0, stream>>>(tb, ln1_g + l * 256, ln1_b + l * 256, zb);

    // FF
    gemm_kernel<<<dim3(16, 8, 1), 256, 0, stream>>>(
        zb, 256, 0, 0, wff1 + (long)l * 262144, 1024, 0, 0, 0, bff1 + l * 1024, 1,
        nullptr, 0, fb, 1024, 0, 0, nullptr, 0, 1024, 1024, 256, 1.f, 1, 0);
    gemm_kernel<<<dim3(4, 8, 1), 256, 0, stream>>>(
        fb, 1024, 0, 0, wff2 + (long)l * 262144, 256, 0, 0, 0, bff2 + l * 256, 1,
        zb, 0, tb, 256, 0, 0, nullptr, 0, 1024, 256, 1024, 1.f, 0, 0);
    ln_kernel<<<1024, 256, 0, stream>>>(tb, ln2_g + l * 256, ln2_b + l * 256, zb);
  }

  // bilinear head: zw = z @ head_w ; logits[b] = zw_b @ z_b^T + head_b
  gemm_kernel<<<dim3(4, 8, 1), 256, 0, stream>>>(
      zb, 256, 0, 0, head_w, 256, 0, 0, 0, nullptr, 0, nullptr, 0,
      zwb, 256, 0, 0, nullptr, 0, 1024, 256, 256, 1.f, 0, 0);
  gemm_kernel<<<dim3(2, 1, 8), 256, 0, stream>>>(
      zwb, 256, 32768, 0, zb, 256, 32768, 0, 1, head_b, 2, nullptr, 0,
      out, 128, 16384, 0, nullptr, 0, 128, 128, 256, 1.f, 0, 0);
}
